// SegmentAttention_78228534329369
// MI455X (gfx1250) — matrix operand order repaired and checked by reading, not compiled, not run on hardware
//
#include <hip/hip_runtime.h>

// ---------------- problem constants ----------------
#define B_     16
#define NQ_    100
#define C_     256
#define HW_    1024     // 32*32
#define P_     16384    // 128*128
#define HEADS_ 8
#define HD_    32
#define DFF_   1024
#define SCALE_ 0.17677669529663687f
#define EPS_   1e-5f

// ---------------- WMMA types ----------------
typedef __attribute__((ext_vector_type(16))) __bf16 v16bf;
typedef __attribute__((ext_vector_type(8)))  float  v8f;

union BF16x16 {
  __bf16 h[16];
  uint4  q[2];    // two 16-byte halves for raw bf16 loads
  v16bf  v;
};

// ============================================================
// fp32 -> bf16 bulk conversion (8 elements / thread)
// ============================================================
__global__ void conv_bf16_kernel(const float* __restrict__ in,
                                 unsigned short* __restrict__ out, long n)
{
  long i = ((long)blockIdx.x * blockDim.x + threadIdx.x) * 8;
  if (i < n) {
    union { __bf16 h[8]; uint4 u; } o;
#pragma unroll
    for (int e = 0; e < 8; ++e) o.h[e] = (__bf16)in[i + e];
    *(uint4*)(out + i) = o.u;
  }
}

// ============================================================
// Generic batched GEMM:  C[b] = act(A[b] (MxK) @ W[b] (KxN) + bias)
// A fp32 (AB16=false) or bf16 (AB16=true); W likewise via WB16.
// bf16 WMMA compute, fp32 accumulate.
// Block = 256 threads = 8 waves; wave computes 16(M) x (NT*16)(N);
// waves stacked along M -> block tile 128 x NT*16.
// Requires: K % 32 == 0, N % (NT*16) == 0.
// ============================================================
template<bool AB16, bool WB16, int NT>
__global__ void gemm_wmma(const void* __restrict__ A, long strideA, int lda,
                          const void* __restrict__ W, long strideW, int ldw,
                          const float* __restrict__ bias,
                          float* __restrict__ C, long strideC, int ldc,
                          int M, int K, int act)
{
  const int b = blockIdx.z;
  float* Cb = C + (long)b * strideC;
  const float*          Af = (const float*)A          + (AB16 ? 0 : (long)b * strideA);
  const unsigned short* Ah = (const unsigned short*)A + (AB16 ? (long)b * strideA : 0);
  const float*          Wf = (const float*)W          + (WB16 ? 0 : (long)b * strideW);
  const unsigned short* Wh = (const unsigned short*)W + (WB16 ? (long)b * strideW : 0);

  const int lane = threadIdx.x & 31;
  const int wave = threadIdx.x >> 5;
  const int half = lane >> 4;       // 0: lanes 0-15, 1: lanes 16-31
  const int l16  = lane & 15;

  const int mbase = blockIdx.y * 128 + wave * 16;
  const int n0    = blockIdx.x * (NT * 16);

  int arow = mbase + l16;
  if (arow >= M) arow = M - 1;      // clamp: junk rows computed, never stored

  v8f zero = {};
  v8f acc[NT];
#pragma unroll
  for (int t = 0; t < NT; ++t) acc[t] = zero;

  for (int k0 = 0; k0 < K; k0 += 32) {
    // A fragment: lane holds row (l16); element e -> K = (e/8)*16 + half*8 + (e%8)
    BF16x16 au;
    if (AB16) {
      const unsigned short* ap = Ah + (long)arow * lda + k0 + half * 8;
      au.q[0] = *(const uint4*)(ap);
      au.q[1] = *(const uint4*)(ap + 16);
    } else {
      const float* arp = Af + (long)arow * lda + k0 + half * 8;
#pragma unroll
      for (int e = 0; e < 8; ++e) au.h[e]     = (__bf16)arp[e];
#pragma unroll
      for (int e = 0; e < 8; ++e) au.h[8 + e] = (__bf16)arp[16 + e];
    }
    const v16bf af = au.v;

    // B fragment: lane = N column (l16); element e -> K = (e/8)*16 + half*8 + (e%8)
#pragma unroll
    for (int t = 0; t < NT; ++t) {
      BF16x16 bu;
      const int bcol = n0 + t * 16 + l16;
      if (WB16) {
#pragma unroll
        for (int e = 0; e < 16; ++e) {
          unsigned short raw = Wh[(long)(k0 + (e >> 3) * 16 + half * 8 + (e & 7)) * ldw + bcol];
          bu.h[e] = __builtin_bit_cast(__bf16, raw);
        }
      } else {
#pragma unroll
        for (int e = 0; e < 16; ++e)
          bu.h[e] = (__bf16)Wf[(long)(k0 + (e >> 3) * 16 + half * 8 + (e & 7)) * ldw + bcol];
      }
      acc[t] = __builtin_amdgcn_wmma_f32_16x16x32_bf16(
          false, af, false, bu.v, (short)0, acc[t], false, false);
    }
  }

  // epilogue: C layout -> lane column n0+t*16+l16 ; VGPR r -> row r + 8*half
#pragma unroll
  for (int t = 0; t < NT; ++t) {
    const int ncol = n0 + t * 16 + l16;
    const float bv = bias ? bias[ncol] : 0.0f;
#pragma unroll
    for (int r = 0; r < 8; ++r) {
      const int m = mbase + r + 8 * half;
      if (m < M) {
        float v = acc[t][r] + bv;
        if (act) v = fmaxf(v, 0.0f);
        Cb[(long)m * ldc + ncol] = v;
      }
    }
  }
}

// ============================================================
// features (B,C,32,32) -> featT (B,HW,C)
// ============================================================
__global__ void transpose_feat_kernel(const float* __restrict__ f, float* __restrict__ ft)
{
  long idx = (long)blockIdx.x * blockDim.x + threadIdx.x;
  if (idx < (long)B_ * HW_ * C_) {
    int c = (int)(idx & (C_ - 1));
    long r = idx >> 8;
    int m = (int)(r & (HW_ - 1));
    int b = (int)(r >> 10);
    ft[idx] = f[((long)b * C_ + c) * HW_ + m];
  }
}

// queries init: broadcast query_embed over batch
__global__ void broadcast_queries_kernel(const float* __restrict__ qe, float* __restrict__ out)
{
  long idx = (long)blockIdx.x * blockDim.x + threadIdx.x;
  if (idx < (long)B_ * NQ_ * C_) out[idx] = qe[idx % ((long)NQ_ * C_)];
}

// ============================================================
// attention mask: sigmoid(mask_pred 128x128) -> antialiased bilinear
// downsample to 32x32 -> (>0.5) -> empty-row fix. One block per (b,n).
// ============================================================
__global__ void attn_mask_kernel(const float* __restrict__ mp, float* __restrict__ am)
{
  const int row = blockIdx.x;                       // b*NQ + n
  const float* m = mp + (long)row * P_;
  float* amrow = am + (long)row * HW_;
  const float w8[8] = {0.125f, 0.375f, 0.625f, 0.875f, 0.875f, 0.625f, 0.375f, 0.125f};

  __shared__ int tot;
  if (threadIdx.x == 0) tot = 0;
  __syncthreads();

  float vals[4];
  int cnt = 0;
#pragma unroll
  for (int q = 0; q < 4; ++q) {
    const int p = threadIdx.x + q * 256;            // 0..1023
    const int oi = p >> 5, oj = p & 31;
    float s = 0.f;
    for (int di = 0; di < 8; ++di) {
      int ii = 4 * oi - 2 + di;
      ii = ii < 0 ? 0 : (ii > 127 ? 127 : ii);
      float rs = 0.f;
      for (int dj = 0; dj < 8; ++dj) {
        int jj = 4 * oj - 2 + dj;
        jj = jj < 0 ? 0 : (jj > 127 ? 127 : jj);
        const float x = m[ii * 128 + jj];
        rs += w8[dj] / (1.0f + __expf(-x));
      }
      s += w8[di] * rs;
    }
    const float bit = (s * (1.0f / 16.0f) > 0.5f) ? 1.0f : 0.0f;
    vals[q] = bit;
    cnt += (bit > 0.f) ? 1 : 0;
  }
  atomicAdd(&tot, cnt);
  __syncthreads();
  const float fix = (tot == 0) ? 1.0f : 0.0f;       // empty rows attend everywhere
#pragma unroll
  for (int q = 0; q < 4; ++q)
    amrow[threadIdx.x + q * 256] = fmaxf(vals[q], fix);
}

// ============================================================
// masked MHA, flash-style online softmax. Block per (b,h), 128 thr.
// ============================================================
__global__ void attn_kernel(const float* __restrict__ qp, const float* __restrict__ kp,
                            const float* __restrict__ vp, const float* __restrict__ am,
                            float* __restrict__ out)
{
  __shared__ float kt[64 * HD_];
  __shared__ float vt[64 * HD_];
  const int b = blockIdx.x / HEADS_;
  const int h = blockIdx.x % HEADS_;
  const int n = threadIdx.x;
  const bool active = (n < NQ_);

  float q[HD_], acc[HD_];
  float mrun = -3.0e38f, lrun = 0.f;
#pragma unroll
  for (int d = 0; d < HD_; ++d) acc[d] = 0.f;
  if (active) {
#pragma unroll
    for (int d = 0; d < HD_; ++d)
      q[d] = qp[((long)b * NQ_ + n) * C_ + h * HD_ + d];
  }
  const float* amrow = am + ((long)b * NQ_ + (active ? n : 0)) * HW_;

  for (int m0 = 0; m0 < HW_; m0 += 64) {
    __syncthreads();
    for (int idx = threadIdx.x; idx < 64 * HD_; idx += blockDim.x) {
      const int j = idx / HD_, d = idx % HD_;
      const long g = ((long)b * HW_ + m0 + j) * C_ + h * HD_ + d;
      kt[idx] = kp[g];
      vt[idx] = vp[g];
    }
    __syncthreads();
    if (active) {
      for (int j = 0; j < 64; ++j) {
        if (amrow[m0 + j] > 0.5f) {
          float s = 0.f;
#pragma unroll
          for (int d = 0; d < HD_; ++d) s += q[d] * kt[j * HD_ + d];
          s *= SCALE_;
          const float mnew = fmaxf(mrun, s);
          const float corr = __expf(mrun - mnew);
          const float p    = __expf(s - mnew);
          lrun = lrun * corr + p;
#pragma unroll
          for (int d = 0; d < HD_; ++d)
            acc[d] = acc[d] * corr + p * vt[j * HD_ + d];
          mrun = mnew;
        }
      }
    }
  }
  if (active) {
    const float inv = 1.0f / fmaxf(lrun, 1e-30f);
#pragma unroll
    for (int d = 0; d < HD_; ++d)
      out[((long)b * NQ_ + n) * C_ + h * HD_ + d] = acc[d] * inv;
  }
}

// ============================================================
// out = LayerNorm(x (+ res)) * g + b ; one block (256 thr) per row.
// ============================================================
__global__ void add_layernorm_kernel(const float* __restrict__ x, const float* __restrict__ res,
                                     const float* __restrict__ g, const float* __restrict__ bta,
                                     float* __restrict__ out, float* __restrict__ out2)
{
  const int row = blockIdx.x;
  const int t = threadIdx.x;
  float v = x[(long)row * C_ + t];
  if (res) v += res[(long)row * C_ + t];
  __shared__ float s1[256], s2[256];
  s1[t] = v; s2[t] = v * v;
  __syncthreads();
  for (int off = 128; off > 0; off >>= 1) {
    if (t < off) { s1[t] += s1[t + off]; s2[t] += s2[t + off]; }
    __syncthreads();
  }
  const float mean = s1[0] * (1.0f / C_);
  const float var  = s2[0] * (1.0f / C_) - mean * mean;
  const float o = (v - mean) * rsqrtf(var + EPS_) * g[t] + bta[t];
  out[(long)row * C_ + t] = o;
  if (out2) out2[(long)row * C_ + t] = o;
}

// ============================================================
// host side
// ============================================================
extern "C" void kernel_launch(void* const* d_in, const int* in_sizes, int n_in,
                              void* d_out, int out_size, void* d_ws, size_t ws_size,
                              hipStream_t stream)
{
  (void)out_size;
  const float* features = (const float*)d_in[0];
  const float* maskfeat = (const float*)d_in[1];

  struct Lp { const float *wq,*bq,*wk,*bk,*wv,*bv,*wo,*bo,*n1g,*n1b,*fw1,*fb1,*fw2,*fb2,*n2g,*n2b; } L[2];
  const float *qe, *fng, *fnb, *mw1, *mb1, *mw2, *mb2;
  auto F = [&](int i) { return (const float*)d_in[i]; };

  if (n_in >= 41 && in_sizes[2] == NQ_ * C_) {
    // insertion-order flatten
    int i = 2; qe = F(i++);
    for (int l = 0; l < 2; ++l) {
      L[l].wq = F(i++); L[l].bq = F(i++); L[l].wk = F(i++); L[l].bk = F(i++);
      L[l].wv = F(i++); L[l].bv = F(i++); L[l].wo = F(i++); L[l].bo = F(i++);
      L[l].n1g = F(i++); L[l].n1b = F(i++); L[l].fw1 = F(i++); L[l].fb1 = F(i++);
      L[l].fw2 = F(i++); L[l].fb2 = F(i++); L[l].n2g = F(i++); L[l].n2b = F(i++);
    }
    fng = F(i++); fnb = F(i++);
    mw1 = F(i++); mb1 = F(i++); mw2 = F(i++); mb2 = F(i++);
  } else {
    // jax tree_flatten (sorted keys)
    int i = 2; fnb = F(i++); fng = F(i++);
    for (int l = 0; l < 2; ++l) {
      L[l].bk = F(i++); L[l].bo = F(i++); L[l].bq = F(i++); L[l].bv = F(i++);
      L[l].fb1 = F(i++); L[l].fb2 = F(i++); L[l].fw1 = F(i++); L[l].fw2 = F(i++);
      L[l].n1b = F(i++); L[l].n1g = F(i++); L[l].n2b = F(i++); L[l].n2g = F(i++);
      L[l].wk = F(i++); L[l].wo = F(i++); L[l].wq = F(i++); L[l].wv = F(i++);
    }
    mb1 = F(i++); mb2 = F(i++); mw1 = F(i++); mw2 = F(i++);
    qe = F(i++);
  }

  // workspace carving
  char* ws = (char*)d_ws;
  size_t cur = 0;
  auto alloc = [&](size_t bytes) -> void* {
    void* p = (void*)(ws + cur);
    cur += (bytes + 255) & ~(size_t)255;
    return p;
  };
  float* featT    = (float*)alloc((size_t)B_ * HW_ * C_ * 4);
  float* queries  = (float*)alloc((size_t)B_ * NQ_ * C_ * 4);
  float* e1       = (float*)alloc((size_t)B_ * NQ_ * C_ * 4);
  float* e2       = (float*)alloc((size_t)B_ * NQ_ * C_ * 4);
  float* qp       = (float*)alloc((size_t)B_ * NQ_ * C_ * 4);
  float* attn_out = (float*)alloc((size_t)B_ * NQ_ * C_ * 4);
  float* proj     = (float*)alloc((size_t)B_ * NQ_ * C_ * 4);
  float* ffn1     = (float*)alloc((size_t)B_ * NQ_ * DFF_ * 4);
  float* kp       = (float*)alloc((size_t)B_ * HW_ * C_ * 4);
  float* vp       = (float*)alloc((size_t)B_ * HW_ * C_ * 4);
  float* am       = (float*)alloc((size_t)B_ * NQ_ * HW_ * 4);
  unsigned short* e2b = (unsigned short*)alloc((size_t)B_ * NQ_ * C_ * 2);

  // optional bf16 copy of mask_features (eliminates in-loop cvt VALU)
  const size_t mf_elems = (size_t)B_ * C_ * P_;
  const bool use_bf16_mf = (cur + mf_elems * 2 <= ws_size);
  unsigned short* mfb = use_bf16_mf ? (unsigned short*)alloc(mf_elems * 2) : nullptr;

  float* outf = (float*)d_out;
  const long QOFF  = 0;
  const long PMOFF = (long)B_ * NQ_ * C_;                  // 409600
  const long IOFF  = PMOFF + (long)B_ * NQ_ * P_;          // 26624000
  const long MSZ   = (long)B_ * NQ_ * P_;                  // per-layer mask block

  // small GEMMs: fp32 A and W, 16x64 per wave
  auto gemmF = [&](const float* A, long sA, int lda, const float* W, long sW, int ldw,
                   const float* bias, float* Cc, long sC, int ldc,
                   int M, int K, int N, int act) {
    dim3 grid((unsigned)(N / 64), (unsigned)((M + 127) / 128), (unsigned)B_);
    gemm_wmma<false, false, 4><<<grid, 256, 0, stream>>>(
        (const void*)A, sA, lda, (const void*)W, sW, ldw, bias, Cc, sC, ldc, M, K, act);
  };
  // mask-prediction big GEMM: e2 @ mask_features -> dst  (16x128 per wave)
  auto mask_gemm = [&](float* dst) {
    dim3 grid((unsigned)(P_ / 128), 1u, (unsigned)B_);
    if (use_bf16_mf) {
      // pre-convert A (e2) to bf16 -> no in-loop cvt; A via b128 loads, B via 16-bit loads
      long n = (long)B_ * NQ_ * C_;
      conv_bf16_kernel<<<(unsigned)((n / 8 + 255) / 256), 256, 0, stream>>>(e2, e2b, n);
      gemm_wmma<true, true, 8><<<grid, 256, 0, stream>>>(
          (const void*)e2b, (long)NQ_ * C_, C_, (const void*)mfb, (long)C_ * P_, P_,
          nullptr, dst, (long)NQ_ * P_, P_, NQ_, C_, 0);
    } else {
      gemm_wmma<false, false, 8><<<grid, 256, 0, stream>>>(
          (const void*)e2, (long)NQ_ * C_, C_, (const void*)maskfeat, (long)C_ * P_, P_,
          nullptr, dst, (long)NQ_ * P_, P_, NQ_, C_, 0);
    }
  };

  {
    if (use_bf16_mf) {
      long nblk = (long)(mf_elems / 8 + 255) / 256;
      conv_bf16_kernel<<<(unsigned)nblk, 256, 0, stream>>>(maskfeat, mfb, (long)mf_elems);
    }
    long n = (long)B_ * HW_ * C_;
    transpose_feat_kernel<<<(unsigned)((n + 255) / 256), 256, 0, stream>>>(features, featT);
    long nq = (long)B_ * NQ_ * C_;
    broadcast_queries_kernel<<<(unsigned)((nq + 255) / 256), 256, 0, stream>>>(qe, queries);
  }

  for (int l = 0; l < 2; ++l) {
    // ---- predict_mask(queries) -> intermediate slot l ----
    gemmF(queries, (long)NQ_ * C_, C_, mw1, 0, C_, mb1, e1, (long)NQ_ * C_, C_, NQ_, C_, C_, 1);
    gemmF(e1,      (long)NQ_ * C_, C_, mw2, 0, C_, mb2, e2, (long)NQ_ * C_, C_, NQ_, C_, C_, 0);
    float* inter = outf + IOFF + (long)l * MSZ;
    mask_gemm(inter);

    // ---- attention mask from mask_pred ----
    attn_mask_kernel<<<B_ * NQ_, 256, 0, stream>>>(inter, am);

    // ---- projections ----
    gemmF(queries, (long)NQ_ * C_, C_, L[l].wq, 0, C_, L[l].bq, qp, (long)NQ_ * C_, C_, NQ_, C_, C_, 0);
    gemmF(featT,   (long)HW_ * C_, C_, L[l].wk, 0, C_, L[l].bk, kp, (long)HW_ * C_, C_, HW_, C_, C_, 0);
    gemmF(featT,   (long)HW_ * C_, C_, L[l].wv, 0, C_, L[l].bv, vp, (long)HW_ * C_, C_, HW_, C_, C_, 0);

    // ---- masked attention ----
    attn_kernel<<<B_ * HEADS_, 128, 0, stream>>>(qp, kp, vp, am, attn_out);

    // ---- output projection + residual LN ----
    gemmF(attn_out, (long)NQ_ * C_, C_, L[l].wo, 0, C_, L[l].bo, proj, (long)NQ_ * C_, C_, NQ_, C_, C_, 0);
    add_layernorm_kernel<<<B_ * NQ_, 256, 0, stream>>>(proj, queries, L[l].n1g, L[l].n1b, queries, nullptr);

    // ---- FFN + residual LN ----
    gemmF(queries, (long)NQ_ * C_, C_, L[l].fw1, 0, DFF_, L[l].fb1, ffn1, (long)NQ_ * DFF_, DFF_, NQ_, C_, DFF_, 1);
    gemmF(ffn1, (long)NQ_ * DFF_, DFF_, L[l].fw2, 0, C_, L[l].fb2, proj, (long)NQ_ * C_, C_, NQ_, DFF_, C_, 0);
    add_layernorm_kernel<<<B_ * NQ_, 256, 0, stream>>>(proj, queries, L[l].n2g, L[l].n2b, queries, nullptr);
  }

  // ---- final norm (also writes queries output slot) ----
  add_layernorm_kernel<<<B_ * NQ_, 256, 0, stream>>>(queries, nullptr, fng, fnb, queries, outf + QOFF);

  // ---- final predict_mask -> pred_masks slot ----
  gemmF(queries, (long)NQ_ * C_, C_, mw1, 0, C_, mb1, e1, (long)NQ_ * C_, C_, NQ_, C_, C_, 1);
  gemmF(e1,      (long)NQ_ * C_, C_, mw2, 0, C_, mb2, e2, (long)NQ_ * C_, C_, NQ_, C_, C_, 0);
  mask_gemm(outf + PMOFF);
}
